// MultiHeadAttention_17678085391133
// MI455X (gfx1250) — matrix operand order repaired and checked by reading, not compiled, not run on hardware
//
#include <hip/hip_runtime.h>
#include <math.h>

// ---------------- types ----------------
typedef __attribute__((ext_vector_type(16))) __bf16 v16bf;
typedef __attribute__((ext_vector_type(8)))  __bf16 v8bf;
typedef __attribute__((ext_vector_type(8)))  float  v8f;
typedef __attribute__((ext_vector_type(4)))  int    v4i;

// ---- CDNA5 async global->LDS DMA (ASYNCcnt path), guarded so we always compile ----
#if defined(__HIP_DEVICE_COMPILE__) && defined(__has_builtin)
#  if __has_builtin(__builtin_amdgcn_global_load_async_to_lds_b128)
#    define HAVE_ASYNC_LDS 1
#  endif
#endif
#ifndef HAVE_ASYNC_LDS
#  define HAVE_ASYNC_LDS 0
#endif

__device__ __forceinline__ void cp16_async(const __bf16* g, __bf16* l) {
#if HAVE_ASYNC_LDS
    typedef __attribute__((address_space(1))) v4i gv4i;   // global int4*
    typedef __attribute__((address_space(3))) v4i lv4i;   // LDS int4*
    __builtin_amdgcn_global_load_async_to_lds_b128((gv4i*)g, (lv4i*)l, 0, 0);
    __builtin_amdgcn_global_load_async_to_lds_b128((gv4i*)(g + 8), (lv4i*)(l + 8), 0, 0);
#else
    *(v8bf*)l       = *(const v8bf*)g;
    *(v8bf*)(l + 8) = *(const v8bf*)(g + 8);
#endif
}

__device__ __forceinline__ void wait_async() {
#if HAVE_ASYNC_LDS
    asm volatile("s_wait_asynccnt 0" ::: "memory");
#endif
}

__device__ __forceinline__ v8f wmma_bf16(v16bf a, v16bf b, v8f c) {
    // D = A(16x32 bf16) * B(32x16 bf16) + C(16x16 f32)
    return __builtin_amdgcn_wmma_f32_16x16x32_bf16(
        false, a, false, b, (short)0, c, false, false);
}

__device__ __forceinline__ v8bf cvt8(float4 a, float4 b) {
    v8bf r;
    r[0] = (__bf16)a.x; r[1] = (__bf16)a.y; r[2] = (__bf16)a.z; r[3] = (__bf16)a.w;
    r[4] = (__bf16)b.x; r[5] = (__bf16)b.y; r[6] = (__bf16)b.z; r[7] = (__bf16)b.w;
    return r;
}

__device__ __forceinline__ v16bf combine16(v8bf lo, v8bf hi8) {
    v16bf r;
#pragma unroll
    for (int i = 0; i < 8; ++i) { r[i] = lo[i]; r[i + 8] = hi8[i]; }
    return r;
}

__device__ __forceinline__ v16bf ld16(const __bf16* p) {
    return combine16(*(const v8bf*)p, *(const v8bf*)(p + 16));
}

// A-fragment (16x32, MxK): lane m = lane&15; K elements {8hi..8hi+7} U {16+8hi..16+8hi+7}
__device__ __forceinline__ v16bf loadA16(const __bf16* p, int hi) {
    v8bf l = *(const v8bf*)(p + 8 * hi);
    v8bf h = *(const v8bf*)(p + 16 + 8 * hi);
    return combine16(l, h);
}

// ---------------- fp32 -> bf16 demotion (bandwidth-bound, one-time) ----------------
__global__ __launch_bounds__(256) void cvt_f32_to_bf16(const float* __restrict__ in,
                                                       __bf16* __restrict__ out, int n8) {
    const int i = blockIdx.x * 256 + threadIdx.x;
    if (i >= n8) return;
    const float4* p = (const float4*)in + (size_t)2 * i;
    ((v8bf*)out)[i] = cvt8(p[0], p[1]);
}

// ---------------- GEMM: C[m,n] = sum_k A[m,k] * B[n,k]  (TN, all-bf16) ----------------
// block = 256 threads (8 waves); block tile 128(M) x 128(N); per-wave 32x64; K step 32.
// Double-buffered LDS staged with async global->LDS DMA.
template <typename CT>
__global__ __launch_bounds__(256) void gemm_bf16_wmma(const __bf16* __restrict__ A,
                                                      const __bf16* __restrict__ B,
                                                      CT* __restrict__ C,
                                                      int M, int N, int K) {
    constexpr int LD = 40;  // 32 + 8 pad -> 80B row stride, conflict-free frag reads
    __shared__ __bf16 Al[2][128 * LD];
    __shared__ __bf16 Bl[2][128 * LD];

    const int tid = threadIdx.x, lane = tid & 31, w = tid >> 5;
    const int nl = lane & 15, hi = lane >> 4;
    const int mi = (w & 3) * 32, ni = (w >> 2) * 64;   // wave sub-tile in block tile
    const int m0 = blockIdx.x * 128, n0 = blockIdx.y * 128;

    // staging: thread -> (row 0..127, k-half 0/16); 2x b128 per matrix per step
    const int srow = tid >> 1, soff = (tid & 1) * 16;
    const __bf16* Ag = A + (size_t)(m0 + srow) * K + soff;
    const __bf16* Bg = B + (size_t)(n0 + srow) * K + soff;
    const int ldst = srow * LD + soff;

    v8f acc[2][4] = {};

    cp16_async(Ag, &Al[0][ldst]);   // prologue: stage k=0 into buffer 0
    cp16_async(Bg, &Bl[0][ldst]);

    int cur = 0;
    for (int k = 0; k < K; k += 32) {
        wait_async();               // own async writes landed
        __syncthreads();            // everyone's writes visible / prior reads done
        if (k + 32 < K) {           // prefetch next tile into the other buffer
            cp16_async(Ag + k + 32, &Al[cur ^ 1][ldst]);
            cp16_async(Bg + k + 32, &Bl[cur ^ 1][ldst]);
        }
        v16bf af[2];
#pragma unroll
        for (int s = 0; s < 2; ++s) {
            const __bf16* ap = &Al[cur][(mi + s * 16 + nl) * LD];
            af[s] = combine16(*(const v8bf*)(ap + 8 * hi),
                              *(const v8bf*)(ap + 16 + 8 * hi));
        }
#pragma unroll
        for (int f = 0; f < 4; ++f) {
            v16bf bfr = ld16(&Bl[cur][(ni + f * 16 + nl) * LD + 8 * hi]);
            acc[0][f] = wmma_bf16(af[0], bfr, acc[0][f]);
            acc[1][f] = wmma_bf16(af[1], bfr, acc[1][f]);
        }
        cur ^= 1;
    }

    // C-frag: element r -> row (tile_m + r + 8*hi), col (tile_n + lane&15)
#pragma unroll
    for (int s = 0; s < 2; ++s)
#pragma unroll
        for (int f = 0; f < 4; ++f)
#pragma unroll
            for (int r = 0; r < 8; ++r)
                C[(size_t)(m0 + mi + s * 16 + r + 8 * hi) * N +
                  n0 + ni + f * 16 + nl] = static_cast<CT>(acc[s][f][r]);
}

// ---------------- Flash attention forward (causal) ----------------
// grid: (S/64, H, B); block 128 (4 waves x 16 q-rows).
__global__ __launch_bounds__(128) void attn_fwd(const __bf16* __restrict__ qkv,
                                                __bf16* __restrict__ aout) {
    constexpr int S   = 2048;
    constexpr int DM  = 2048;
    constexpr int LDQ = 3 * DM;      // 6144 token stride in qkv buffer
    constexpr int LDK = 136;         // K tile row pad
    constexpr int LDV = 40;          // Vt / P row pad
    __shared__ __bf16 Kl[32 * LDK];       // [kv][d]   32 x 128
    __shared__ __bf16 Vt[128 * LDV];      // [d][kv]   128 x 32 (transposed)
    __shared__ __bf16 Pl[4][16 * LDV];    // per-wave P tile 16 x 32

    const int tid  = threadIdx.x;
    const int w    = tid >> 5;
    const int lane = tid & 31;
    const int nl   = lane & 15;
    const int hi   = lane >> 4;

    const int qb = blockIdx.x, h = blockIdx.y, b = blockIdx.z;
    const int q0 = qb * 64 + w * 16;

    const size_t tokbase = (size_t)b * S * LDQ;
    const __bf16* Qp = qkv + tokbase + h * 128;            // c=0
    const __bf16* Kp = qkv + tokbase + DM + h * 128;       // c=1
    const __bf16* Vp = qkv + tokbase + 2 * DM + h * 128;   // c=2

    // preload Q fragments (16 x 128 = 4 A-frags of 16x32)
    v16bf qa[4];
    {
        const __bf16* qrow = Qp + (size_t)(q0 + nl) * LDQ;
#pragma unroll
        for (int kd = 0; kd < 4; ++kd) qa[kd] = loadA16(qrow + kd * 32, hi);
    }

    v8f o[8] = {};
    float mrow[8], lrow[8];
#pragma unroll
    for (int r = 0; r < 8; ++r) { mrow[r] = -1e30f; lrow[r] = 0.f; }
    const float sc = 0.08838834764831845f * 1.44269504088896f;  // 1/sqrt(128)*log2(e)

    const int krow = tid >> 2, kseg = tid & 3;   // K staging: 32 rows x 4 segs of 32
    const int vrow = tid & 31, vseg = tid >> 5;  // V staging: kv row, d seg (0..3)

    const int ntiles = qb * 2 + 2;               // kv up to block's max q, step 32
    for (int t = 0; t < ntiles; ++t) {
        const int c0 = t * 32;
        __syncthreads();
        {   // stage K tile row-major via async DMA to LDS
            const __bf16* src = Kp + (size_t)(c0 + krow) * LDQ + kseg * 32;
            __bf16* dst = &Kl[krow * LDK + kseg * 32];
            cp16_async(src, dst);
            cp16_async(src + 16, dst + 16);
        }
        {   // stage V tile transposed: Vt[d][kv]
            const __bf16* vs = Vp + (size_t)(c0 + vrow) * LDQ + vseg * 32;
            v8bf vv[4];
#pragma unroll
            for (int c = 0; c < 4; ++c) vv[c] = *(const v8bf*)(vs + c * 8);
#pragma unroll
            for (int j = 0; j < 32; ++j)
                Vt[(vseg * 32 + j) * LDV + vrow] = vv[j / 8][j % 8];
        }
        wait_async();
        __syncthreads();

        // scores: S = Q K^T  (two 16x16 C-frags covering 32 kv columns)
        v8f s2[2] = {};
#pragma unroll
        for (int halfn = 0; halfn < 2; ++halfn)
#pragma unroll
            for (int kd = 0; kd < 4; ++kd) {
                const __bf16* kp = &Kl[(halfn * 16 + nl) * LDK + kd * 32 + 8 * hi];
                s2[halfn] = wmma_bf16(qa[kd], ld16(kp), s2[halfn]);
            }

        // scale + causal mask + per-row online softmax
        float sv[2][8], rm[8];
        const bool diag = (c0 + 31 > q0);   // wave-uniform
#pragma unroll
        for (int r = 0; r < 8; ++r) {
            const int q = q0 + r + 8 * hi;
            float v0 = s2[0][r] * sc, v1 = s2[1][r] * sc;
            if (diag) {
                if (c0 + nl > q)      v0 = -1e30f;
                if (c0 + 16 + nl > q) v1 = -1e30f;
            }
            sv[0][r] = v0; sv[1][r] = v1;
            rm[r] = fmaxf(v0, v1);
        }
#pragma unroll
        for (int r = 0; r < 8; ++r)
#pragma unroll
            for (int off = 1; off < 16; off <<= 1)
                rm[r] = fmaxf(rm[r], __shfl_xor(rm[r], off, 16));

        float ps[8], al[8];
#pragma unroll
        for (int r = 0; r < 8; ++r) {
            const float mnew = fmaxf(mrow[r], rm[r]);
            al[r] = exp2f(mrow[r] - mnew);
            mrow[r] = mnew;
            const float p0 = exp2f(sv[0][r] - mnew);
            const float p1 = exp2f(sv[1][r] - mnew);
            sv[0][r] = p0; sv[1][r] = p1;
            ps[r] = p0 + p1;
#pragma unroll
            for (int f = 0; f < 8; ++f) o[f][r] *= al[r];
        }
#pragma unroll
        for (int r = 0; r < 8; ++r) {
#pragma unroll
            for (int off = 1; off < 16; off <<= 1)
                ps[r] += __shfl_xor(ps[r], off, 16);
            lrow[r] = lrow[r] * al[r] + ps[r];
        }

        // re-layout P (C-frag -> A-frag) through per-wave LDS
        __bf16* pw = &Pl[w][0];
#pragma unroll
        for (int halfn = 0; halfn < 2; ++halfn)
#pragma unroll
            for (int r = 0; r < 8; ++r)
                pw[(r + 8 * hi) * LDV + halfn * 16 + nl] = (__bf16)sv[halfn][r];
        asm volatile("s_wait_dscnt 0" ::: "memory");  // wave-local RAW through LDS
        v16bf pa;
        {
            const __bf16* pr = &Pl[w][nl * LDV + 8 * hi];
            pa = combine16(*(const v8bf*)pr, *(const v8bf*)(pr + 16));
        }

        // O += P * V  (8 d-chunks of 16)
#pragma unroll
        for (int f = 0; f < 8; ++f) {
            const __bf16* vp = &Vt[(f * 16 + nl) * LDV + 8 * hi];
            o[f] = wmma_bf16(pa, ld16(vp), o[f]);
        }
    }

    // epilogue: normalize and store bf16 in [token, h*128 + d] layout
#pragma unroll
    for (int r = 0; r < 8; ++r) {
        const float inv = 1.0f / lrow[r];
        const size_t tok = (size_t)b * S + q0 + r + 8 * hi;
#pragma unroll
        for (int f = 0; f < 8; ++f)
            aout[tok * DM + h * 128 + f * 16 + nl] = (__bf16)(o[f][r] * inv);
    }
}

// ---------------- host launch ----------------
extern "C" void kernel_launch(void* const* d_in, const int* in_sizes, int n_in,
                              void* d_out, int out_size, void* d_ws, size_t ws_size,
                              hipStream_t stream) {
    (void)in_sizes; (void)n_in; (void)out_size; (void)ws_size;
    const float* x    = (const float*)d_in[0];   // [2, 2048, 2048]
    const float* Wqkv = (const float*)d_in[1];   // [6144, 2048]
    const float* Wout = (const float*)d_in[2];   // [2048, 2048]
    float* out = (float*)d_out;                  // [2, 2048, 2048]

    // workspace layout (bf16): x, Wqkv, Wout demoted copies + qkv + attn out
    __bf16* xb    = (__bf16*)d_ws;                       // [4096, 2048]
    __bf16* wqkvb = xb    + (size_t)4096 * 2048;         // [6144, 2048]
    __bf16* woutb = wqkvb + (size_t)6144 * 2048;         // [2048, 2048]
    __bf16* qkv   = woutb + (size_t)2048 * 2048;         // [4096, 6144]
    __bf16* aout  = qkv   + (size_t)4096 * 6144;         // [4096, 2048]

    dim3 b256(256), b128(128);
    // 0) demote fp32 inputs to bf16 (one-time, ~10us of HBM traffic)
    cvt_f32_to_bf16<<<(4096 * 2048 / 8) / 256, b256, 0, stream>>>(x, xb, 4096 * 2048 / 8);
    cvt_f32_to_bf16<<<(6144 * 2048 / 8) / 256, b256, 0, stream>>>(Wqkv, wqkvb, 6144 * 2048 / 8);
    cvt_f32_to_bf16<<<(2048 * 2048 / 8) / 256, b256, 0, stream>>>(Wout, woutb, 2048 * 2048 / 8);
    // 1) QKV projection: qkv = x @ Wqkv^T   (M=4096, N=6144, K=2048)
    gemm_bf16_wmma<__bf16><<<dim3(4096 / 128, 6144 / 128), b256, 0, stream>>>(
        xb, wqkvb, qkv, 4096, 6144, 2048);
    // 2) causal flash attention
    attn_fwd<<<dim3(2048 / 64, 16, 2), b128, 0, stream>>>(qkv, aout);
    // 3) output projection: out = aout @ Wout^T  (M=4096, N=2048, K=2048)
    gemm_bf16_wmma<float><<<dim3(4096 / 128, 2048 / 128), b256, 0, stream>>>(
        aout, woutb, out, 4096, 2048, 2048);
}
